// GraphSAGE_Net_21784074125386
// MI455X (gfx1250) — matrix operand order repaired and checked by reading, not compiled, not run on hardware
//
#include <hip/hip_runtime.h>
#include <stdint.h>

// ---------------------------------------------------------------------------
// Types for CDNA5 WMMA (wave32): v_wmma_f32_16x16x32_bf16
// ---------------------------------------------------------------------------
typedef __attribute__((ext_vector_type(16))) __bf16 v16bf;
typedef __attribute__((ext_vector_type(8)))  float  v8f;

__device__ __forceinline__ unsigned short f2bf(float f) {
    unsigned u = __builtin_bit_cast(unsigned, f);
    u += 0x7FFFu + ((u >> 16) & 1u);           // round-to-nearest-even
    return (unsigned short)(u >> 16);
}

// ---------------------------------------------------------------------------
// Edge-degree count:  cnt[dst] += 1
// ---------------------------------------------------------------------------
__global__ __launch_bounds__(256)
void sage_count(const long long* __restrict__ ei, float* __restrict__ cnt, int nE) {
    int e = blockIdx.x * 256 + threadIdx.x;
    if (e < nE) atomicAdd(&cnt[(int)ei[(size_t)nE + e]], 1.0f);
}

// ---------------------------------------------------------------------------
// Scatter-add of feature rows along edges: agg[dst] += feat[src]
// One wave (32 lanes) per edge; VEC float4's per lane (F = 128*VEC).
// ---------------------------------------------------------------------------
template<int VEC>
__global__ __launch_bounds__(256)
void sage_scatter(const long long* __restrict__ ei, const float* __restrict__ feat,
                  float* __restrict__ agg, int nE, int F) {
    long long gid = (long long)blockIdx.x * 256 + threadIdx.x;
    int edge = (int)(gid >> 5);
    int lane = (int)(gid & 31);
    if (edge >= nE) return;
    int s = (int)ei[edge];
    int d = (int)ei[(size_t)nE + edge];
    const float* __restrict__ srow = feat + (size_t)s * F;
    float* __restrict__ drow = agg + (size_t)d * F;
#pragma unroll
    for (int c = 0; c < VEC; ++c) {
        int q = (lane + 32 * c) * 4;
        float4 v = *(const float4*)(srow + q);
        atomicAdd(drow + q + 0, v.x);
        atomicAdd(drow + q + 1, v.y);
        atomicAdd(drow + q + 2, v.z);
        atomicAdd(drow + q + 3, v.w);
    }
}

// ---------------------------------------------------------------------------
// Build bf16 A operand: Abf[row][0:KHALF) = agg[row]/max(cnt,1),
//                       Abf[row][KHALF:2*KHALF) = xr[row]; zero pad rows >= M.
// 8 bf16 (one uint4) per thread.
// ---------------------------------------------------------------------------
template<int KHALF>
__global__ __launch_bounds__(256)
void sage_convert(const float* __restrict__ agg, const float* __restrict__ cnt,
                  const float* __restrict__ xr, unsigned short* __restrict__ Abf,
                  int M, int Mpad) {
    constexpr int KTOT = 2 * KHALF;
    constexpr int SEGS = KTOT / 8;
    long long unit = (long long)blockIdx.x * 256 + threadIdx.x;
    int row = (int)(unit / SEGS);
    int k0  = (int)(unit % SEGS) * 8;
    if (row >= Mpad) return;
    union { unsigned short s[8]; uint4 u; } pk;
    if (row < M) {
        if (k0 < KHALF) {
            float inv = 1.0f / fmaxf(cnt[row], 1.0f);
            const float* p = agg + (size_t)row * KHALF + k0;
#pragma unroll
            for (int j = 0; j < 8; ++j) pk.s[j] = f2bf(p[j] * inv);
        } else {
            const float* p = xr + (size_t)row * KHALF + (k0 - KHALF);
#pragma unroll
            for (int j = 0; j < 8; ++j) pk.s[j] = f2bf(p[j]);
        }
    } else {
#pragma unroll
        for (int j = 0; j < 8; ++j) pk.s[j] = 0;
    }
    *(uint4*)&Abf[(size_t)row * KTOT + k0] = pk.u;
}

// ---------------------------------------------------------------------------
// Build bf16 weight operand, transposed+stacked: Wt[n][k] (k-major rows),
// k < KHALF -> Wl[k][n], else Wr[k-KHALF][n]; zero for n >= NCOLS.
// ---------------------------------------------------------------------------
template<int KHALF, int NCOLS, int NPAD>
__global__ __launch_bounds__(256)
void sage_prepw(const float* __restrict__ Wl, const float* __restrict__ Wr,
                unsigned short* __restrict__ Wt) {
    constexpr int KTOT = 2 * KHALF;
    int idx = blockIdx.x * 256 + threadIdx.x;
    if (idx >= NPAD * KTOT) return;
    int n = idx / KTOT, k = idx % KTOT;
    float v = 0.0f;
    if (n < NCOLS)
        v = (k < KHALF) ? Wl[(size_t)k * NCOLS + n]
                        : Wr[(size_t)(k - KHALF) * NCOLS + n];
    Wt[idx] = f2bf(v);
}

// ---------------------------------------------------------------------------
// SAGE GEMM:  out = act( Abf @ Wt^T + bias )   (bf16 WMMA, f32 accum)
//   Abf: Mpad x KTOT bf16 row-major;  Wt: NPAD x KTOT bf16 (n-major rows)
// Block: 64 rows x NPAD cols, 256 threads (8 waves), K-step 32.
// A tile staged via GLOBAL_LOAD_ASYNC_TO_LDS_B128 (ASYNCcnt); B fragments
// loaded directly from global (L2-resident weights).
// ---------------------------------------------------------------------------
template<int KHALF, int NCOLS, int NPAD, bool RELU>
__global__ __launch_bounds__(256)
void sage_gemm(const unsigned short* __restrict__ Abf,
               const unsigned short* __restrict__ Wt,
               const float* __restrict__ bias, float* __restrict__ out, int M) {
    constexpr int CT   = NPAD / 16;        // column tiles per block
    constexpr int TPW  = (4 * CT) / 8;     // 16x16 tiles per wave
    constexpr int KTOT = 2 * KHALF;

    __shared__ __align__(16) unsigned short As[64][40];   // [m][k], +8 pad

    const int t    = threadIdx.x;
    const int wave = t >> 5;
    const int lane = t & 31;
    const int row0 = blockIdx.x * 64;

    // One 16B async chunk per thread covers the 64x32 A tile.
    const int ar = t >> 2;                 // row within tile
    const int ac = (t & 3) * 8;            // k offset within 32
    const unsigned lds_dst = (unsigned)(uintptr_t)&As[ar][ac];
    const unsigned short* gsrc = Abf + (size_t)(row0 + ar) * KTOT + ac;

    const int half = (lane >> 4) & 1;
    const int hkA  = half * 8;             // A frag: K {0..7,16..23}/{8..15,24..31}
    const int kbB  = half * 8;             // B frag: K {0..7,16..23}/{8..15,24..31}
    const int mrow = lane & 15;
    const int rtw  = (wave * TPW) / CT;    // row-tile (uniform per wave)

    v8f acc[TPW];
#pragma unroll
    for (int i = 0; i < TPW; ++i)
#pragma unroll
        for (int j = 0; j < 8; ++j) acc[i][j] = 0.0f;

    for (int kk = 0; kk < KTOT; kk += 32) {
        __syncthreads();   // previous iteration done reading As
        // ---- async copy A tile (global -> LDS), ASYNCcnt-tracked ----
        {
            unsigned long long ga = (unsigned long long)(uintptr_t)(gsrc + kk);
            asm volatile("global_load_async_to_lds_b128 %0, %1, off"
                         :: "v"(lds_dst), "v"(ga) : "memory");
        }
        // ---- B fragments straight from global while async is in flight ----
        union { uint4 u[2]; v16bf v; } bfm[TPW];
#pragma unroll
        for (int i = 0; i < TPW; ++i) {
            int ct  = (wave * TPW + i) % CT;
            const unsigned short* wp =
                Wt + (size_t)(ct * 16 + mrow) * KTOT + kk + kbB;
            bfm[i].u[0] = *(const uint4*)(wp);
            bfm[i].u[1] = *(const uint4*)(wp + 16);
        }
        asm volatile("s_wait_asynccnt 0" ::: "memory");
        __syncthreads();   // A tile visible to all waves
        // ---- A fragment (shared by this wave's tiles) + WMMA ----
        union { uint4 u[2]; v16bf v; } af;
        af.u[0] = *(const uint4*)&As[rtw * 16 + mrow][hkA];
        af.u[1] = *(const uint4*)&As[rtw * 16 + mrow][hkA + 16];
#pragma unroll
        for (int i = 0; i < TPW; ++i) {
            acc[i] = __builtin_amdgcn_wmma_f32_16x16x32_bf16(
                false, af.v, false, bfm[i].v, (short)0, acc[i], false, false);
        }
    }

    // ---- epilogue: bias (+ReLU), masked store ----
    const int mofs = half * 8;
    const int ncol = lane & 15;
#pragma unroll
    for (int i = 0; i < TPW; ++i) {
        int ct  = (wave * TPW + i) % CT;
        int col = ct * 16 + ncol;
#pragma unroll
        for (int r = 0; r < 8; ++r) {
            int grow = row0 + rtw * 16 + mofs + r;
            if (grow < M && col < NCOLS) {
                float v = acc[i][r] + bias[col];
                if (RELU) v = fmaxf(v, 0.0f);
                out[(size_t)grow * NCOLS + col] = v;
            }
        }
    }
}

// ---------------------------------------------------------------------------
// Row-wise log_softmax over 40 classes, in place. One wave per node.
// ---------------------------------------------------------------------------
__global__ __launch_bounds__(256)
void sage_logsoftmax(float* __restrict__ out, int M) {
    int wid  = (blockIdx.x * 256 + threadIdx.x) >> 5;
    int lane = threadIdx.x & 31;
    if (wid >= M) return;
    float* row = out + (size_t)wid * 40;
    float v0 = row[lane];
    float v1 = (lane < 8) ? row[32 + lane] : -3.4e38f;
    float mx = fmaxf(v0, v1);
#pragma unroll
    for (int m = 16; m >= 1; m >>= 1) mx = fmaxf(mx, __shfl_xor(mx, m, 32));
    float s = __expf(v0 - mx) + ((lane < 8) ? __expf(v1 - mx) : 0.0f);
#pragma unroll
    for (int m = 16; m >= 1; m >>= 1) s += __shfl_xor(s, m, 32);
    float lg = __logf(s);
    row[lane] = v0 - mx - lg;
    if (lane < 8) row[32 + lane] = v1 - mx - lg;
}

// ---------------------------------------------------------------------------
// Host-side launcher
// ---------------------------------------------------------------------------
extern "C" void kernel_launch(void* const* d_in, const int* in_sizes, int n_in,
                              void* d_out, int out_size, void* d_ws, size_t ws_size,
                              hipStream_t stream) {
    constexpr int FIN = 128, FHID = 256, FOUT = 40;

    const float*     x   = (const float*)d_in[0];
    const long long* ei  = (const long long*)d_in[1];   // int64 [2, E]
    const float*     W1l = (const float*)d_in[2];
    const float*     W1r = (const float*)d_in[3];
    const float*     b1  = (const float*)d_in[4];
    const float*     W2l = (const float*)d_in[5];
    const float*     W2r = (const float*)d_in[6];
    const float*     b2  = (const float*)d_in[7];
    float*           y   = (float*)d_out;

    const int N    = in_sizes[0] / FIN;
    const int nE   = in_sizes[1] / 2;
    const int Mpad = (N + 63) & ~63;

    auto alg = [](size_t b) { return (b + 255) & ~(size_t)255; };
    const size_t szAgg1 = alg((size_t)N * FIN * sizeof(float));
    const size_t szAbf1 = alg((size_t)Mpad * (2 * FIN) * sizeof(unsigned short));
    const size_t szAbf2 = alg((size_t)Mpad * (2 * FHID) * sizeof(unsigned short));
    const size_t szR0   = (szAgg1 + szAbf1 > szAbf2) ? (szAgg1 + szAbf1) : szAbf2;

    char* w = (char*)d_ws;
    // Region R0: agg1 + Abf1 live in layer 1; Abf2 aliases it in layer 2.
    float*          agg1 = (float*)w;
    unsigned short* Abf1 = (unsigned short*)(w + szAgg1);
    unsigned short* Abf2 = (unsigned short*)w;
    w += szR0;
    float* cnt  = (float*)w;  w += alg((size_t)N * sizeof(float));
    float* h    = (float*)w;  w += alg((size_t)N * FHID * sizeof(float));
    float* agg2 = (float*)w;  w += alg((size_t)N * FHID * sizeof(float));
    unsigned short* Wt1 = (unsigned short*)w;
    w += alg((size_t)256 * (2 * FIN) * sizeof(unsigned short));
    unsigned short* Wt2 = (unsigned short*)w;
    w += alg((size_t)64 * (2 * FHID) * sizeof(unsigned short));

    hipMemsetAsync(agg1, 0, (size_t)N * FIN  * sizeof(float), stream);
    hipMemsetAsync(cnt,  0, (size_t)N        * sizeof(float), stream);
    hipMemsetAsync(agg2, 0, (size_t)N * FHID * sizeof(float), stream);

    sage_count<<<(nE + 255) / 256, 256, 0, stream>>>(ei, cnt, nE);

    const int sblk = (int)(((long long)nE * 32 + 255) / 256);
    const int gblk = Mpad / 64;

    // ---- Layer 1:  K = 2*128, N = 256, ReLU ----
    sage_scatter<1><<<sblk, 256, 0, stream>>>(ei, x, agg1, nE, FIN);
    sage_convert<FIN><<<((size_t)Mpad * (2 * FIN / 8) + 255) / 256, 256, 0, stream>>>(
        agg1, cnt, x, Abf1, N, Mpad);
    sage_prepw<FIN, FHID, 256><<<(256 * 2 * FIN + 255) / 256, 256, 0, stream>>>(
        W1l, W1r, Wt1);
    sage_gemm<FIN, FHID, 256, true><<<gblk, 256, 0, stream>>>(Abf1, Wt1, b1, h, N);

    // ---- Layer 2:  K = 2*256, N = 40 (padded to 64) ----
    sage_scatter<2><<<sblk, 256, 0, stream>>>(ei, h, agg2, nE, FHID);
    sage_convert<FHID><<<((size_t)Mpad * (2 * FHID / 8) + 255) / 256, 256, 0, stream>>>(
        agg2, cnt, h, Abf2, N, Mpad);
    sage_prepw<FHID, FOUT, 64><<<(64 * 2 * FHID + 255) / 256, 256, 0, stream>>>(
        W2l, W2r, Wt2);
    sage_gemm<FHID, FOUT, 64, false><<<gblk, 256, 0, stream>>>(Abf2, Wt2, b2, y, N);

    // ---- log_softmax in place on d_out ----
    sage_logsoftmax<<<((size_t)N * 32 + 255) / 256, 256, 0, stream>>>(y, N);
}
